// BVRNN_43104291782657
// MI455X (gfx1250) — matrix operand order repaired and checked by reading, not compiled, not run on hardware
//
#include <hip/hip_runtime.h>
#include <hip/hip_bf16.h>
#include <math.h>

typedef __bf16 bf16;
typedef bf16  v16bf __attribute__((ext_vector_type(16)));
typedef bf16  v8bf  __attribute__((ext_vector_type(8)));
typedef float v8f   __attribute__((ext_vector_type(8)));

#define EPS_K 1e-4f

// ---------------- device helpers ----------------

__device__ __forceinline__ float sigm(float v) { return 1.0f / (1.0f + __expf(-v)); }

template <int ACT>
__device__ __forceinline__ float act_fn(float v) {
    if constexpr (ACT == 1) {                    // ELU
        return v > 0.0f ? v : (__expf(v) - 1.0f);
    } else if constexpr (ACT == 2 || ACT == 3) { // sigmoid (3 = sigmoid, round on bf16 path)
        return 1.0f / (1.0f + __expf(-v));
    } else {
        return v;
    }
}

// A fragment (16x32 bf16, M x K): lane L<16 -> row L, K = kb..kb+15;
// lanes 16..31 -> row L-16, K shifted by +16. Two 16B loads + shuffle.
__device__ __forceinline__ v16bf load_a_frag(const bf16* __restrict__ arow, int kb) {
    v8bf lo = *(const v8bf*)(arow + kb);
    v8bf hi = *(const v8bf*)(arow + kb + 8);
    return __builtin_shufflevector(lo, hi, 0, 1, 2, 3, 4, 5, 6, 7,
                                           8, 9, 10, 11, 12, 13, 14, 15);
}

// ---------------- WMMA GEMM with fused bias + activation ----------------
// out[m,n] = act( sum_k A[m,k] * W[n,k] + bias[n] )   (A = [A1 | A2] along K)
// One wave computes a 16 x 64 strip: one A-fragment feeds NT=4 *unconditional*
// independent v_wmma_f32_16x16x32_bf16 chains (W is N-padded to a multiple of
// 64 rows with zeros at conversion time, so no guards touch the WMMA loop).
// Wave/tile indices go through readfirstlane so all control flow is scalar.

template <int ACT>
__global__ __launch_bounds__(256) void wmma_gemm(
    const bf16* __restrict__ A1, int lda1, int K1,
    const bf16* __restrict__ A2, int lda2, int K2,
    const bf16* __restrict__ W, int ldw,
    const float* __restrict__ bias,
    float* __restrict__ outf, int ldo,
    bf16* __restrict__ outb, int ldob,
    int M, int N)
{
    constexpr int NT = 4;
    const int lane = threadIdx.x & 31;
    // Force wave id into an SGPR: all derived control flow is scalar (no EXEC churn).
    const int wave = __builtin_amdgcn_readfirstlane(
        blockIdx.x * (blockDim.x >> 5) + (threadIdx.x >> 5));
    const int mtiles = M >> 4;
    const int ngroups = (((N + 15) >> 4) + NT - 1) / NT;  // W padded to ngroups*64 rows
    if (wave >= mtiles * ngroups) return;                 // scalar branch
    const int mt = wave % mtiles;
    const int nt0 = (wave / mtiles) * NT;

    const int half = lane >> 4;
    const int lrow = lane & 15;

    const bf16* a1row = A1 + (long)(mt * 16 + lrow) * lda1 + half * 16;
    const bf16* a2row = (K2 > 0) ? (A2 + (long)(mt * 16 + lrow) * lda2 + half * 16) : A1;

    // B operand: W row n supplies column n of B (out = x @ W^T); 16 contiguous K per lane.
    const bf16* wrow[NT];
#pragma unroll
    for (int j = 0; j < NT; ++j)
        wrow[j] = W + (long)((nt0 + j) * 16 + lrow) * ldw + half * 16;

    v8f acc[NT] = {};

    for (int kb = 0; kb < K1; kb += 32) {
        v16bf a = load_a_frag(a1row, kb);
        if (kb + 32 < K1) {
#pragma unroll
            for (int j = 0; j < NT; ++j)
                __builtin_prefetch((const void*)(wrow[j] + kb + 32), 0, 1);
        }
#pragma unroll
        for (int j = 0; j < NT; ++j)
            acc[j] = __builtin_amdgcn_wmma_f32_16x16x32_bf16(
                false, a, false, *(const v16bf*)(wrow[j] + kb),
                (short)0, acc[j], false, false);
    }
    for (int kb = 0; kb < K2; kb += 32) {
        v16bf a = load_a_frag(a2row, kb);
        if (kb + 32 < K2) {
#pragma unroll
            for (int j = 0; j < NT; ++j)
                __builtin_prefetch((const void*)(wrow[j] + K1 + kb + 32), 0, 1);
        }
#pragma unroll
        for (int j = 0; j < NT; ++j)
            acc[j] = __builtin_amdgcn_wmma_f32_16x16x32_bf16(
                false, a, false, *(const v16bf*)(wrow[j] + K1 + kb),
                (short)0, acc[j], false, false);
    }

    // C/D layout: VGPR r -> row (mt*16 + r + 8*half), col (nt*16 + (lane&15)).
    // Only stores are guarded for the N-padded tail (per-lane, no WMMA involved).
    const int rbase = mt * 16 + (half << 3);
#pragma unroll
    for (int j = 0; j < NT; ++j) {
        const int col = (nt0 + j) * 16 + lrow;
        const bool colok = col < N;
        const float bv = (bias && colok) ? bias[col] : 0.0f;
#pragma unroll
        for (int r = 0; r < 8; ++r) {
            float v = acc[j][r] + bv;
            float a = act_fn<ACT>(v);
            int row = rbase + r;
            if (outf && colok) outf[(long)row * ldo + col] = a;
            if (outb && colok) {
                float s = (ACT == 3) ? rintf(a) : a;  // straight-through round (RNE)
                outb[(long)row * ldob + col] = (bf16)s;
            }
        }
    }
}

// ---------------- elementwise kernels ----------------

// yn = (y - mean)/std, written bf16 into a K-padded [rows, 96] buffer (cols 80..95 = 0)
__global__ void normalize_pad_kernel(const float* __restrict__ y,
                                     const float* __restrict__ meanm,
                                     const float* __restrict__ stdm,
                                     bf16* __restrict__ yn, int total /* rows*96 */)
{
    int i = blockIdx.x * blockDim.x + threadIdx.x;
    if (i >= total) return;
    int r = i / 96, x = i - r * 96;
    float v = 0.0f;
    if (x < 80) v = (y[r * 80 + x] - meanm[x]) / stdm[x];
    yn[i] = (bf16)v;
}

// fp32 [N,K] weight -> bf16 [Np,Kp] (zero-padded along both N and K)
__global__ void cvt_weight_kernel(const float* __restrict__ W, bf16* __restrict__ Wb,
                                  int K, int Kp, int N, int total /* Np*Kp */)
{
    int i = blockIdx.x * blockDim.x + threadIdx.x;
    if (i >= total) return;
    int n = i / Kp, k = i - n * Kp;
    float v = (n < N && k < K) ? W[(long)n * K + k] : 0.0f;
    Wb[i] = (bf16)v;
}

__global__ void zero_state_kernel(float* __restrict__ h, bf16* __restrict__ hb, int n)
{
    int i = blockIdx.x * blockDim.x + threadIdx.x;
    if (i >= n) return;
    h[i] = 0.0f;
    hb[i] = (bf16)0.0f;
}

// GRU cell combine: r,z,n gates from gi/gh [B,3H]; updates h fp32 and bf16 copy
__global__ void gru_combine_kernel(const float* __restrict__ gi,
                                   const float* __restrict__ gh,
                                   float* __restrict__ h, bf16* __restrict__ hb,
                                   int H, int total /* B*H */)
{
    int i = blockIdx.x * blockDim.x + threadIdx.x;
    if (i >= total) return;
    int b = i / H, j = i - b * H;
    const float* gib = gi + (long)b * 3 * H;
    const float* ghb = gh + (long)b * 3 * H;
    float r = sigm(gib[j] + ghb[j]);
    float z = sigm(gib[H + j] + ghb[H + j]);
    float n = tanhf(gib[2 * H + j] + r * ghb[2 * H + j]);
    float hn = (1.0f - z) * n + z * h[i];
    h[i] = hn;
    hb[i] = (bf16)hn;
}

// Per-step KLD: mean over batch of sum over Z -> klds[t] (single block: deterministic)
__global__ void kld_step_kernel(const float* __restrict__ enc,
                                const float* __restrict__ prior,
                                float* __restrict__ kout)
{
    __shared__ float red[256];
    float s = 0.0f;
    for (int i = threadIdx.x; i < 64 * 64; i += 256) {
        float e = enc[i], p = prior[i];
        s += e * (logf(e + EPS_K) - logf(p + EPS_K))
           + (1.0f - e) * (logf(1.0f - e + EPS_K) - logf(1.0f - p + EPS_K));
    }
    red[threadIdx.x] = s;
    __syncthreads();
    for (int off = 128; off > 0; off >>= 1) {
        if (threadIdx.x < off) red[threadIdx.x] += red[threadIdx.x + off];
        __syncthreads();
    }
    if (threadIdx.x == 0) *kout = red[0] * (1.0f / 64.0f);
}

__global__ void kld_final_kernel(const float* __restrict__ klds, int T,
                                 float* __restrict__ out)
{
    __shared__ float red[256];
    float s = 0.0f;
    for (int i = threadIdx.x; i < T; i += 256) s += klds[i];
    red[threadIdx.x] = s;
    __syncthreads();
    for (int off = 128; off > 0; off >>= 1) {
        if (threadIdx.x < off) red[threadIdx.x] += red[threadIdx.x + off];
        __syncthreads();
    }
    if (threadIdx.x == 0) *out = red[0] / (float)T;
}

// ---------------- host orchestration ----------------

extern "C" void kernel_launch(void* const* d_in, const int* in_sizes, int n_in,
                              void* d_out, int out_size, void* d_ws, size_t ws_size,
                              hipStream_t stream)
{
    (void)in_sizes; (void)n_in; (void)out_size; (void)ws_size;

    constexpr int B = 64, T = 500, X = 80, H = 512, Z = 64;
    constexpr int ROWS = B * T;          // 32000
    constexpr int XP = 96;               // X padded to multiple of 32 (K dim)
    constexpr int XNP = 128;             // X padded to multiple of 64 (N dim, dec output)

    // --- inputs (setup_inputs dict order; params in insertion order, W then b) ---
    const float* y     = (const float*)d_in[0];
    const float* meanm = (const float*)d_in[1];
    const float* stdm  = (const float*)d_in[2];
    const float* Wx0 = (const float*)d_in[3];  const float* bx0 = (const float*)d_in[4];
    const float* Wx1 = (const float*)d_in[5];  const float* bx1 = (const float*)d_in[6];
    const float* Wx2 = (const float*)d_in[7];  const float* bx2 = (const float*)d_in[8];
    const float* Wz0 = (const float*)d_in[9];  const float* bz0 = (const float*)d_in[10];
    const float* Wz1 = (const float*)d_in[11]; const float* bz1 = (const float*)d_in[12];
    const float* Wz2 = (const float*)d_in[13]; const float* bz2 = (const float*)d_in[14];
    const float* We0 = (const float*)d_in[15]; const float* be0 = (const float*)d_in[16];
    const float* We1 = (const float*)d_in[17]; const float* be1 = (const float*)d_in[18];
    const float* We2 = (const float*)d_in[19]; const float* be2 = (const float*)d_in[20];
    const float* Wp0 = (const float*)d_in[21]; const float* bp0 = (const float*)d_in[22];
    const float* Wp1 = (const float*)d_in[23]; const float* bp1 = (const float*)d_in[24];
    const float* Wp2 = (const float*)d_in[25]; const float* bp2 = (const float*)d_in[26];
    const float* Wd0 = (const float*)d_in[27]; const float* bd0 = (const float*)d_in[28];
    const float* Wd1 = (const float*)d_in[29]; const float* bd1 = (const float*)d_in[30];
    const float* Wd2 = (const float*)d_in[31]; const float* bd2 = (const float*)d_in[32];
    const float* Wd3 = (const float*)d_in[33]; const float* bd3 = (const float*)d_in[34];
    const float* Wih = (const float*)d_in[35]; const float* Whh = (const float*)d_in[36];
    const float* bih = (const float*)d_in[37]; const float* bhh = (const float*)d_in[38];

    float* dout = (float*)d_out;
    float* kld_out = dout + (size_t)B * T * X;

    // --- workspace bump allocator ---
    char* ws = (char*)d_ws;
    size_t off = 0;
    auto alloc = [&](size_t bytes) -> void* {
        void* p = ws + off;
        off = (off + bytes + 255) & ~(size_t)255;
        return p;
    };
    auto abf = [&](size_t elems) { return (bf16*)alloc(elems * sizeof(bf16)); };
    auto af  = [&](size_t elems) { return (float*)alloc(elems * sizeof(float)); };

    // bf16 weights (~11 MB total -> L2 resident across all 500 steps)
    bf16* Wx0b = abf((size_t)H * XP);
    bf16* Wx1b = abf((size_t)H * H);
    bf16* Wx2b = abf((size_t)H * H);
    bf16* Wz0b = abf((size_t)H * Z);
    bf16* Wz1b = abf((size_t)H * H);
    bf16* Wz2b = abf((size_t)H * H);
    bf16* We0b = abf((size_t)H * 2 * H);
    bf16* We1b = abf((size_t)H * H);
    bf16* We2b = abf((size_t)Z * H);
    bf16* Wp0b = abf((size_t)H * H);
    bf16* Wp1b = abf((size_t)H * H);
    bf16* Wp2b = abf((size_t)Z * H);
    bf16* Wd0b = abf((size_t)H * 2 * H);
    bf16* Wd1b = abf((size_t)H * H);
    bf16* Wd2b = abf((size_t)H * H);
    bf16* Wd3b = abf((size_t)XNP * H);   // N-padded 80 -> 128 rows (zeros)
    bf16* Wihb = abf((size_t)3 * H * 2 * H);
    bf16* Whhb = abf((size_t)3 * H * H);

    // activations
    bf16* yn   = abf((size_t)ROWS * XP);
    bf16* bufA = abf((size_t)ROWS * H);
    bf16* bufB = abf((size_t)ROWS * H);
    bf16* px   = abf((size_t)ROWS * H);   // phi_x output, bf16, [B*T, H]
    float* h   = af((size_t)B * H);
    bf16* hb   = abf((size_t)B * H);
    bf16* e1b  = abf((size_t)B * H);
    bf16* e2b  = abf((size_t)B * H);
    float* encf = af((size_t)B * Z);
    bf16* zb   = abf((size_t)B * Z);
    bf16* p1b  = abf((size_t)B * H);
    bf16* p2b  = abf((size_t)B * H);
    float* priorf = af((size_t)B * Z);
    bf16* z1b  = abf((size_t)B * H);
    bf16* z2b  = abf((size_t)B * H);
    bf16* phizb = abf((size_t)B * H);
    bf16* d1b  = abf((size_t)B * H);
    bf16* d2b  = abf((size_t)B * H);
    bf16* d3b  = abf((size_t)B * H);
    float* gi  = af((size_t)B * 3 * H);
    float* gh  = af((size_t)B * 3 * H);
    float* klds = af((size_t)T);

    // src fp32 [N,K] -> dst bf16 [Np,Kp], zero-padded
    auto cvt = [&](const float* src, bf16* dst, int N, int Np, int K, int Kp) {
        int total = Np * Kp;
        cvt_weight_kernel<<<(total + 255) / 256, 256, 0, stream>>>(src, dst, K, Kp, N, total);
    };
    cvt(Wx0, Wx0b, H, H, X, XP);
    cvt(Wx1, Wx1b, H, H, H, H);      cvt(Wx2, Wx2b, H, H, H, H);
    cvt(Wz0, Wz0b, H, H, Z, Z);      cvt(Wz1, Wz1b, H, H, H, H);  cvt(Wz2, Wz2b, H, H, H, H);
    cvt(We0, We0b, H, H, 2 * H, 2 * H); cvt(We1, We1b, H, H, H, H); cvt(We2, We2b, Z, Z, H, H);
    cvt(Wp0, Wp0b, H, H, H, H);      cvt(Wp1, Wp1b, H, H, H, H);  cvt(Wp2, Wp2b, Z, Z, H, H);
    cvt(Wd0, Wd0b, H, H, 2 * H, 2 * H); cvt(Wd1, Wd1b, H, H, H, H); cvt(Wd2, Wd2b, H, H, H, H);
    cvt(Wd3, Wd3b, X, XNP, H, H);
    cvt(Wih, Wihb, 3 * H, 3 * H, 2 * H, 2 * H); cvt(Whh, Whhb, 3 * H, 3 * H, H, H);

    auto gemm = [&](int act,
                    const bf16* A1, int lda1, int K1,
                    const bf16* A2, int lda2, int K2,
                    const bf16* W, int ldw, const float* bias,
                    float* outf, int ldo, bf16* outb, int ldob, int M, int N) {
        int mtiles = M >> 4;
        int ngroups = (((N + 15) >> 4) + 3) / 4;       // NT = 4 n-tiles per wave
        int blocks = (mtiles * ngroups + 7) / 8;       // 8 waves per 256-thread block
        switch (act) {
        case 0: wmma_gemm<0><<<blocks, 256, 0, stream>>>(A1, lda1, K1, A2, lda2, K2, W, ldw, bias, outf, ldo, outb, ldob, M, N); break;
        case 1: wmma_gemm<1><<<blocks, 256, 0, stream>>>(A1, lda1, K1, A2, lda2, K2, W, ldw, bias, outf, ldo, outb, ldob, M, N); break;
        case 2: wmma_gemm<2><<<blocks, 256, 0, stream>>>(A1, lda1, K1, A2, lda2, K2, W, ldw, bias, outf, ldo, outb, ldob, M, N); break;
        default: wmma_gemm<3><<<blocks, 256, 0, stream>>>(A1, lda1, K1, A2, lda2, K2, W, ldw, bias, outf, ldo, outb, ldob, M, N); break;
        }
    };

    // --- phi_x over all B*T rows (parallel prefix, 3 fused GEMM+ELU) ---
    {
        int total = ROWS * XP;
        normalize_pad_kernel<<<(total + 255) / 256, 256, 0, stream>>>(y, meanm, stdm, yn, total);
    }
    gemm(1, yn, XP, XP, nullptr, 0, 0, Wx0b, XP, bx0, nullptr, 0, bufA, H, ROWS, H);
    gemm(1, bufA, H, H, nullptr, 0, 0, Wx1b, H, bx1, nullptr, 0, bufB, H, ROWS, H);
    gemm(1, bufB, H, H, nullptr, 0, 0, Wx2b, H, bx2, nullptr, 0, px, H, ROWS, H);

    zero_state_kernel<<<(B * H + 255) / 256, 256, 0, stream>>>(h, hb, B * H);

    // --- sequential scan: 500 steps, each a DAG of fused WMMA GEMMs ---
    for (int t = 0; t < T; ++t) {
        const bf16* pxt = px + (size_t)t * H;      // row stride T*H selects batch rows

        // enc: concat(px_t, h) -> H (ELU) -> H (ELU) -> Z (sigmoid; bf16 gets round())
        gemm(1, pxt, T * H, H, hb, H, H, We0b, 2 * H, be0, nullptr, 0, e1b, H, B, H);
        gemm(1, e1b, H, H, nullptr, 0, 0, We1b, H, be1, nullptr, 0, e2b, H, B, H);
        gemm(3, e2b, H, H, nullptr, 0, 0, We2b, H, be2, encf, Z, zb, Z, B, Z);

        // prior: h -> H (ELU) -> H (ELU) -> Z (sigmoid)
        gemm(1, hb, H, H, nullptr, 0, 0, Wp0b, H, bp0, nullptr, 0, p1b, H, B, H);
        gemm(1, p1b, H, H, nullptr, 0, 0, Wp1b, H, bp1, nullptr, 0, p2b, H, B, H);
        gemm(2, p2b, H, H, nullptr, 0, 0, Wp2b, H, bp2, priorf, Z, nullptr, 0, B, Z);

        // phi_z on z = round(enc)
        gemm(1, zb, Z, Z, nullptr, 0, 0, Wz0b, Z, bz0, nullptr, 0, z1b, H, B, H);
        gemm(1, z1b, H, H, nullptr, 0, 0, Wz1b, H, bz1, nullptr, 0, z2b, H, B, H);
        gemm(1, z2b, H, H, nullptr, 0, 0, Wz2b, H, bz2, nullptr, 0, phizb, H, B, H);

        // dec: concat(phi_z, h) -> H -> H -> H (ELU x3) -> X (linear) straight into d_out
        gemm(1, phizb, H, H, hb, H, H, Wd0b, 2 * H, bd0, nullptr, 0, d1b, H, B, H);
        gemm(1, d1b, H, H, nullptr, 0, 0, Wd1b, H, bd1, nullptr, 0, d2b, H, B, H);
        gemm(1, d2b, H, H, nullptr, 0, 0, Wd2b, H, bd2, nullptr, 0, d3b, H, B, H);
        gemm(0, d3b, H, H, nullptr, 0, 0, Wd3b, H, bd3, dout + (size_t)t * X, T * X,
             nullptr, 0, B, X);

        // GRU: gi = concat(px_t, phi_z)@Wih^T + bih ; gh = h@Whh^T + bhh ; combine
        gemm(0, pxt, T * H, H, phizb, H, H, Wihb, 2 * H, bih, gi, 3 * H, nullptr, 0, B, 3 * H);
        gemm(0, hb, H, H, nullptr, 0, 0, Whhb, H, bhh, gh, 3 * H, nullptr, 0, B, 3 * H);
        gru_combine_kernel<<<(B * H + 255) / 256, 256, 0, stream>>>(gi, gh, h, hb, H, B * H);

        kld_step_kernel<<<1, 256, 0, stream>>>(encf, priorf, klds + t);
    }

    kld_final_kernel<<<1, 256, 0, stream>>>(klds, T, kld_out);
}
